// Attention_4896262717567
// MI455X (gfx1250) — matrix operand order repaired and checked by reading, not compiled, not run on hardware
//
#include <hip/hip_runtime.h>
#include <math.h>

// Problem constants
#define SEQ    2048
#define DMODEL 1024
#define NHEAD  16
#define DKH    64
#define SCALE  0.125f   // 1/sqrt(64)

typedef __attribute__((ext_vector_type(16))) __bf16 v16bf;
typedef __attribute__((ext_vector_type(8)))  __bf16 v8bf;
typedef __attribute__((ext_vector_type(8)))  float  v8f;

#define DEV static __device__ __forceinline__

DEV v8f vzero() { v8f z = {0.f,0.f,0.f,0.f,0.f,0.f,0.f,0.f}; return z; }

DEV v16bf cat8(v8bf lo, v8bf hi) {
  return __builtin_shufflevector(lo, hi, 0,1,2,3,4,5,6,7,8,9,10,11,12,13,14,15);
}

// A-fragment: 16x32 bf16 tile, row-major source, leading dim ld (elements).
DEV v16bf load_a_frag(const __bf16* p, int ld) {
  const int lane = threadIdx.x & 31;
  const int m = lane & 15, hi = lane >> 4;
  const __bf16* r = p + m * ld + hi * 8;
  v8bf lo  = *(const v8bf*)(r);
  v8bf hi8 = *(const v8bf*)(r + 16);
  return cat8(lo, hi8);
}

// B-fragment (32x16, B[k][n]) from the TRANSPOSED source T (NxK row-major).
DEV v16bf load_bT_frag(const __bf16* t, int ld) {
  const int lane = threadIdx.x & 31;
  const int n = lane & 15, g = lane >> 4;
  const __bf16* r = t + n * ld + g * 8;
  v8bf lo  = *(const v8bf*)(r);
  v8bf hi8 = *(const v8bf*)(r + 16);
  return cat8(lo, hi8);
}

DEV v8f wmma_bf16(v16bf a, v16bf b, v8f c) {
  return __builtin_amdgcn_wmma_f32_16x16x32_bf16(false, a, false, b, (short)0, c,
                                                 false, false);
}

// Keep all fragment loads ahead of the WMMA group so the wait-insertion pass
// can use partial s_wait_dscnt values instead of recycling one register set.
DEV void sched_fence() { __builtin_amdgcn_sched_barrier(0); }

// ---- CDNA5 async global->LDS copy (per-lane 16B), tracked by ASYNCcnt -------
DEV void async_copy_b128(unsigned lds_off, const __bf16* g) {
  asm volatile("global_load_async_to_lds_b128 %0, %1, off"
               :: "v"(lds_off), "v"((unsigned long long)(size_t)g)
               : "memory");
}
DEV void wait_async0() { asm volatile("s_wait_asynccnt 0x0" ::: "memory"); }

// Generic-pointer low 32 bits == LDS byte offset (aperture lives in [63:32]).
DEV unsigned lds_off_of(const void* p) { return (unsigned)(size_t)p; }

// ---------------------------------------------------------------- convert
__global__ __launch_bounds__(256) void f32_to_bf16_kernel(const float* __restrict__ src,
                                                          __bf16* __restrict__ dst, int n) {
  int i = blockIdx.x * blockDim.x + threadIdx.x;
  const int stride = gridDim.x * blockDim.x;
  for (; i < n; i += stride) dst[i] = (__bf16)src[i];
}

// ---------------------------------------------------------------- shared GEMM core
// Block = 4 waves computing 64x64 of A(row-major) x T^t; K staged 32-deep,
// double-buffered async-to-LDS.  sh elements: A0[0,2048) A1[2048,4096)
// B0[4096,6144) B1[6144,8192).
DEV void gemm_compute_step(const __bf16* As, const __bf16* Bs, v8f* acc) {
  v16bf a  = load_a_frag(As, 32);
  v16bf b0 = load_bT_frag(Bs, 32);
  v16bf b1 = load_bT_frag(Bs + 512, 32);
  v16bf b2 = load_bT_frag(Bs + 1024, 32);
  v16bf b3 = load_bT_frag(Bs + 1536, 32);
  sched_fence();                      // all 10 ds_loads issue before the WMMAs
  acc[0] = wmma_bf16(a, b0, acc[0]);
  acc[1] = wmma_bf16(a, b1, acc[1]);
  acc[2] = wmma_bf16(a, b2, acc[2]);
  acc[3] = wmma_bf16(a, b3, acc[3]);
}

DEV void gemm_mainloop(const __bf16* Ag, const __bf16* Bg, __bf16* sh,
                       v8f* acc, int tid, int wave) {
  const unsigned shOff = lds_off_of(sh);
  // per-thread staging addresses (two 16B chunks per 4KB tile per thread)
  const int r0 = tid >> 2,         o0 = (tid & 3) << 3;
  const int r1 = (tid + 128) >> 2, o1 = (tid & 3) << 3;
  const __bf16* pa0 = Ag + r0 * DMODEL + o0;
  const __bf16* pa1 = Ag + r1 * DMODEL + o1;
  const __bf16* pb0 = Bg + r0 * DMODEL + o0;
  const __bf16* pb1 = Bg + r1 * DMODEL + o1;
  const unsigned da0 = shOff + tid * 16,  da1 = shOff + (tid + 128) * 16;
  const unsigned db0 = da0 + 8192,        db1 = da1 + 8192;

  // prologue: stage k=0 into buffer 0
  async_copy_b128(da0, pa0); async_copy_b128(db0, pb0);
  async_copy_b128(da1, pa1); async_copy_b128(db1, pb1);

  const __bf16* As = sh + wave * 512;
  const __bf16* Bs = sh + 4096;

  #pragma unroll 1
  for (int it = 0; it < 32; it += 2) {
    // even step: compute buffer 0, stage k-step it+1 into buffer 1 (always valid)
    wait_async0();
    __syncthreads();
    {
      const int k1 = (it + 1) << 5;
      async_copy_b128(da0 + 4096, pa0 + k1); async_copy_b128(db0 + 4096, pb0 + k1);
      async_copy_b128(da1 + 4096, pa1 + k1); async_copy_b128(db1 + 4096, pb1 + k1);
    }
    gemm_compute_step(As, Bs, acc);
    // odd step: compute buffer 1, stage k-step it+2 into buffer 0 (skip at end)
    wait_async0();
    __syncthreads();
    if (it + 2 < 32) {
      const int k2 = (it + 2) << 5;
      async_copy_b128(da0, pa0 + k2); async_copy_b128(db0, pb0 + k2);
      async_copy_b128(da1, pa1 + k2); async_copy_b128(db1, pb1 + k2);
    }
    gemm_compute_step(As + 2048, Bs + 2048, acc);
  }
}

// ---------------------------------------------------------------- QKV GEMM
__global__ __launch_bounds__(128) void qkv_gemm_kernel(
    const __bf16* __restrict__ xbf, const __bf16* __restrict__ wab,
    const float* __restrict__ bias,
    __bf16* __restrict__ Qh, __bf16* __restrict__ Kh, __bf16* __restrict__ Vt) {
  __shared__ __align__(16) __bf16 sh[8192];               // 16KB
  const int tid = threadIdx.x, wave = tid >> 5;
  const int mblk = blockIdx.x & 63, nblk = blockIdx.x >> 6;   // 64 x 48
  const int m0 = mblk << 6, n0 = nblk << 6;

  v8f acc[4];
  #pragma unroll
  for (int f = 0; f < 4; ++f) acc[f] = vzero();
  gemm_mainloop(xbf + m0 * DMODEL, wab + n0 * DMODEL, sh, acc, tid, wave);

  const int lane = tid & 31, n = lane & 15, hi = lane >> 4;
  const int m0w = m0 + (wave << 4);
  #pragma unroll
  for (int f = 0; f < 4; ++f) {
    const int e = n0 + f * 16 + n;
    const float bv = bias[e];
    const int part = e >> 10;                             // 0=Q 1=K 2=V
    const int w = e & 1023, h = w >> 6, dk = w & 63;
    #pragma unroll
    for (int i = 0; i < 8; ++i) {
      const int r = m0w + i + (hi << 3);
      const int b = r >> 11, s = r & 2047;
      const int bh = b * NHEAD + h;
      const __bf16 o = (__bf16)(acc[f][i] + bv);
      if (part == 0)      Qh[(bh * SEQ + s) * DKH + dk] = o;
      else if (part == 1) Kh[(bh * SEQ + s) * DKH + dk] = o;
      else                Vt[(bh * DKH + dk) * SEQ + s] = o;
    }
  }
}

// ---------------------------------------------------------------- flash attention
DEV void attn_step(const __bf16* Kt, const __bf16* Vs, __bf16* lp,
                   int j0, int q0, int n, int hi,
                   const v16bf& qa0, const v16bf& qa1,
                   v8f* acc, float* mrun, float* lrun) {
  v16bf kb00 = load_bT_frag(Kt, 64);
  v16bf kb01 = load_bT_frag(Kt + 32, 64);
  v16bf kb10 = load_bT_frag(Kt + 16 * 64, 64);
  v16bf kb11 = load_bT_frag(Kt + 16 * 64 + 32, 64);
  // V fragment loads issued early; consumed after softmax
  v16bf vb0 = load_bT_frag(Vs, 32);
  v16bf vb1 = load_bT_frag(Vs + 512, 32);
  v16bf vb2 = load_bT_frag(Vs + 1024, 32);
  v16bf vb3 = load_bT_frag(Vs + 1536, 32);
  sched_fence();                      // all ds_loads in flight before score WMMAs
  v8f s0 = wmma_bf16(qa0, kb00, vzero());
  s0     = wmma_bf16(qa1, kb01, s0);
  v8f s1 = wmma_bf16(qa0, kb10, vzero());
  s1     = wmma_bf16(qa1, kb11, s1);

  #pragma unroll
  for (int i = 0; i < 8; ++i) {
    const int q = q0 + i + (hi << 3);
    float a = (j0 + n      <= q) ? s0[i] * SCALE : -INFINITY;
    float c = (j0 + 16 + n <= q) ? s1[i] * SCALE : -INFINITY;
    float mx = fmaxf(a, c);
    mx = fmaxf(mx, __shfl_xor(mx, 1));
    mx = fmaxf(mx, __shfl_xor(mx, 2));
    mx = fmaxf(mx, __shfl_xor(mx, 4));
    mx = fmaxf(mx, __shfl_xor(mx, 8));
    const float mnew = fmaxf(mrun[i], mx);
    const float p0 = __expf(a - mnew);
    const float p1 = __expf(c - mnew);
    float rs = p0 + p1;
    rs += __shfl_xor(rs, 1);
    rs += __shfl_xor(rs, 2);
    rs += __shfl_xor(rs, 4);
    rs += __shfl_xor(rs, 8);
    const float corr = __expf(mrun[i] - mnew);
    lrun[i] = lrun[i] * corr + rs;
    mrun[i] = mnew;
    acc[0][i] *= corr; acc[1][i] *= corr; acc[2][i] *= corr; acc[3][i] *= corr;
    lp[(i + (hi << 3)) * 32 + n]      = (__bf16)p0;
    lp[(i + (hi << 3)) * 32 + 16 + n] = (__bf16)p1;
  }
  asm volatile("s_wait_dscnt 0x0" ::: "memory");  // same-wave LDS RAW fence
  v16bf pa = load_a_frag(lp, 32);
  acc[0] = wmma_bf16(pa, vb0, acc[0]);
  acc[1] = wmma_bf16(pa, vb1, acc[1]);
  acc[2] = wmma_bf16(pa, vb2, acc[2]);
  acc[3] = wmma_bf16(pa, vb3, acc[3]);
}

// Block = 4 waves = 4 consecutive 16-row q-tiles of ONE (b,h); K/V tiles shared
// via async-staged, double-buffered LDS. nIter = 2*qg+2 is always even.
__global__ __launch_bounds__(128) void attn_kernel(
    const __bf16* __restrict__ Qh, const __bf16* __restrict__ Kh,
    const __bf16* __restrict__ Vt, __bf16* __restrict__ attn) {
  __shared__ __align__(16) __bf16 shk[2 * 32 * 64];       // 8KB  K tiles [32 keys][64 dk]
  __shared__ __align__(16) __bf16 shv[2 * 64 * 32];       // 8KB  V tiles [64 dk][32 keys]
  __shared__ __align__(16) __bf16 shp[4 * 16 * 32];       // 4KB  per-wave P tiles
  const int tid = threadIdx.x, wave = tid >> 5, lane = tid & 31;
  const int bh = blockIdx.x >> 5, qg = blockIdx.x & 31;   // 32 bh x 32 q-groups
  const int q0 = (qg << 6) + (wave << 4);
  const int n = lane & 15, hi = lane >> 4;
  __bf16* lp = shp + wave * 512;
  const unsigned kOff = lds_off_of(shk), vOff = lds_off_of(shv);
  const __bf16* Kg = Kh + bh * SEQ * DKH;
  const __bf16* Vg = Vt + bh * DKH * SEQ;

  // per-thread staging addresses
  const unsigned dk0 = kOff + tid * 16, dk1 = kOff + (tid + 128) * 16;
  const unsigned dv0 = vOff + tid * 16, dv1 = vOff + (tid + 128) * 16;
  const __bf16* pk0 = Kg + tid * 8;                       // K tile is contiguous 4KB
  const __bf16* pk1 = Kg + (tid + 128) * 8;
  const __bf16* pv0 = Vg + (tid >> 2) * SEQ + ((tid & 3) << 3);
  const __bf16* pv1 = pv0 + 32 * SEQ;                     // rows 32..63

  const __bf16* qbase = Qh + (bh * SEQ + q0) * DKH;
  v16bf qa0 = load_a_frag(qbase, DKH);
  v16bf qa1 = load_a_frag(qbase + 32, DKH);

  v8f acc[4];
  #pragma unroll
  for (int f = 0; f < 4; ++f) acc[f] = vzero();
  float mrun[8], lrun[8];
  #pragma unroll
  for (int i = 0; i < 8; ++i) { mrun[i] = -INFINITY; lrun[i] = 0.f; }

  const int jEnd = q0 + 16;                 // this wave's causal horizon
  const int nIter = (qg << 1) + 2;          // block-max key tiles (uniform, even)

  // prologue: stage key-tile 0 into buffer 0
  async_copy_b128(dk0, pk0); async_copy_b128(dk1, pk1);
  async_copy_b128(dv0, pv0); async_copy_b128(dv1, pv1);

  #pragma unroll 1
  for (int it = 0; it < nIter; it += 2) {
    // even step: compute tile it (buffer 0), stage tile it+1 (always valid)
    wait_async0();
    __syncthreads();
    {
      const int j1 = (it + 1) << 5;
      async_copy_b128(dk0 + 4096, pk0 + j1 * DKH);
      async_copy_b128(dk1 + 4096, pk1 + j1 * DKH);
      async_copy_b128(dv0 + 4096, pv0 + j1);
      async_copy_b128(dv1 + 4096, pv1 + j1);
    }
    if ((it << 5) < jEnd)
      attn_step(shk, shv, lp, it << 5, q0, n, hi, qa0, qa1, acc, mrun, lrun);
    // odd step: compute tile it+1 (buffer 1), stage tile it+2 (skip at end)
    wait_async0();
    __syncthreads();
    if (it + 2 < nIter) {
      const int j2 = (it + 2) << 5;
      async_copy_b128(dk0, pk0 + j2 * DKH);
      async_copy_b128(dk1, pk1 + j2 * DKH);
      async_copy_b128(dv0, pv0 + j2);
      async_copy_b128(dv1, pv1 + j2);
    }
    if (((it + 1) << 5) < jEnd)
      attn_step(shk + 2048, shv + 2048, lp, (it + 1) << 5, q0, n, hi,
                qa0, qa1, acc, mrun, lrun);
  }

  const int b = bh >> 4, h = bh & 15;
  #pragma unroll
  for (int f = 0; f < 4; ++f) {
    #pragma unroll
    for (int i = 0; i < 8; ++i) {
      const int r = q0 + i + (hi << 3);
      attn[(b * SEQ + r) * DMODEL + h * DKH + f * 16 + n] =
          (__bf16)(acc[f][i] / lrun[i]);
    }
  }
}

// ---------------------------------------------------------------- out projection
__global__ __launch_bounds__(128) void proj_gemm_kernel(
    const __bf16* __restrict__ attn, const __bf16* __restrict__ wpb,
    const float* __restrict__ bias, float* __restrict__ out) {
  __shared__ __align__(16) __bf16 sh[8192];
  const int tid = threadIdx.x, wave = tid >> 5;
  const int mblk = blockIdx.x & 63, nblk = blockIdx.x >> 6;   // 64 x 16
  const int m0 = mblk << 6, n0 = nblk << 6;

  v8f acc[4];
  #pragma unroll
  for (int f = 0; f < 4; ++f) acc[f] = vzero();
  gemm_mainloop(attn + m0 * DMODEL, wpb + n0 * DMODEL, sh, acc, tid, wave);

  const int lane = tid & 31, n = lane & 15, hi = lane >> 4;
  const int m0w = m0 + (wave << 4);
  #pragma unroll
  for (int f = 0; f < 4; ++f) {
    const int e = n0 + f * 16 + n;
    const float bv = bias[e];
    #pragma unroll
    for (int i = 0; i < 8; ++i) {
      const int r = m0w + i + (hi << 3);
      out[r * DMODEL + e] = acc[f][i] + bv;
    }
  }
}

// ---------------------------------------------------------------- launch
extern "C" void kernel_launch(void* const* d_in, const int* in_sizes, int n_in,
                              void* d_out, int out_size, void* d_ws, size_t ws_size,
                              hipStream_t stream) {
  (void)in_sizes; (void)n_in; (void)out_size; (void)ws_size;
  const float* x  = (const float*)d_in[0];   // [2,2048,1024]
  const float* wa = (const float*)d_in[1];   // [3072,1024]
  const float* ba = (const float*)d_in[2];   // [3072]
  const float* wp = (const float*)d_in[3];   // [1024,1024]
  const float* bp = (const float*)d_in[4];   // [1024]
  float* out = (float*)d_out;                // [2,2048,1024]

  char* ws = (char*)d_ws;                    // 48 MB used
  __bf16* xbf  = (__bf16*)(ws + 0);          //  8 MB  x bf16
  __bf16* wab  = (__bf16*)(ws + 8388608);    //  6 MB  w_atten bf16
  __bf16* wpb  = (__bf16*)(ws + 14680064);   //  2 MB  w_proj bf16
  __bf16* Qh   = (__bf16*)(ws + 16777216);   //  8 MB  Q [BH][S][64]
  __bf16* Kh   = (__bf16*)(ws + 25165824);   //  8 MB  K [BH][S][64]
  __bf16* Vt   = (__bf16*)(ws + 33554432);   //  8 MB  V^T [BH][64][S]
  __bf16* attn = (__bf16*)(ws + 41943040);   //  8 MB  attn [B][S][D]

  f32_to_bf16_kernel<<<2048, 256, 0, stream>>>(x,  xbf, 2 * SEQ * DMODEL);
  f32_to_bf16_kernel<<<2048, 256, 0, stream>>>(wa, wab, 3 * DMODEL * DMODEL);
  f32_to_bf16_kernel<<<1024, 256, 0, stream>>>(wp, wpb, DMODEL * DMODEL);

  qkv_gemm_kernel<<<3072, 128, 0, stream>>>(xbf, wab, ba, Qh, Kh, Vt);
  attn_kernel<<<1024, 128, 0, stream>>>(Qh, Kh, Vt, attn);
  proj_gemm_kernel<<<1024, 128, 0, stream>>>(attn, wpb, bp, out);
}
